// SurfaceDiceLoss_52776558133771
// MI455X (gfx1250) — compile-verified, matrix operand order checked by reading, not run on hardware
//
#include <hip/hip_runtime.h>

// ---------------- problem constants ----------------
#define H     512
#define W     512
#define NIMG  32
#define TILE  32          // output tile
#define REG   36          // TILE + 2*2 halo (erode radius 1 + dilate radius 1)
#define BND   34          // TILE + 2*1 (boundary grid incl. dilation halo)

typedef __attribute__((ext_vector_type(16))) _Float16 v16h;
typedef __attribute__((ext_vector_type(8)))  float    v8f;
typedef __attribute__((address_space(3))) unsigned char lds_byte_t;

static __device__ __forceinline__ unsigned lds_off(const void* p) {
    // generic -> LDS addrspace -> 32-bit LDS byte address
    return (unsigned)(unsigned long long)(const lds_byte_t*)p;
}

// wave32 sum of a small non-negative integer (exact in f16/f32) using the
// matrix pipe: A has exactly one nonzero element per lane, B = ones, so
// D[m][n] = v[m] + v[m+16]; per-lane sum of the 8 D VGPRs gives half-sums,
// combined with one xor-16 shuffle. Requires EXEC all-ones (uniform call site).
static __device__ __forceinline__ float wave_sum_wmma(float v) {
    v16h a = (v16h)(_Float16)0.0f;
    a[0] = (_Float16)v;
    v16h ones = (v16h)(_Float16)1.0f;
    v8f  c = (v8f)0.0f;
    c = __builtin_amdgcn_wmma_f32_16x16x32_f16(
            /*neg_a=*/false, a, /*neg_b=*/false, ones,
            /*c_mod=*/(short)0, c, /*reuse_a=*/false, /*reuse_b=*/false);
    float s = c[0] + c[1] + c[2] + c[3] + c[4] + c[5] + c[6] + c[7];
    s += __shfl_xor(s, 16, 32);
    return s;
}

__global__ void SurfaceDice_zero(float* acc) {
    acc[threadIdx.x] = 0.0f;   // launched with exactly 96 threads
}

__global__ __launch_bounds__(256)
void SurfaceDice_main(const float* __restrict__ pred,
                      const float* __restrict__ tgt,
                      float* __restrict__ acc) {
    __shared__ float         sf[2][REG * REG];   // staged raw floats (async-to-LDS)
    __shared__ unsigned char msk[2][REG * REG];  // binary masks (0/1)
    __shared__ unsigned char bnd[2][BND * BND];  // boundary (0/1)
    __shared__ float         red[3];             // block partials

    const int tid = threadIdx.x;
    const int b   = blockIdx.z;
    const int y0  = blockIdx.y * TILE;
    const int x0  = blockIdx.x * TILE;

    const unsigned long long pbase = (unsigned long long)pred;
    const unsigned long long tbase = (unsigned long long)tgt;

    // ---- stage 36x36 halo'd tiles of pred/target into LDS via async DMA ----
    // Addresses are clamped into the image so every lane is always valid;
    // out-of-image halo positions are handled when building the masks.
    for (int idx = tid; idx < REG * REG; idx += 256) {
        int r  = idx / REG, c = idx % REG;
        int gy = y0 - 2 + r, gx = x0 - 2 + c;
        int gyc = min(max(gy, 0), H - 1);
        int gxc = min(max(gx, 0), W - 1);
        unsigned goff = (unsigned)((((b * H) + gyc) * W + gxc) * 4);
        unsigned l0 = lds_off(&sf[0][idx]);
        unsigned l1 = lds_off(&sf[1][idx]);
        asm volatile("global_load_async_to_lds_b32 %0, %1, %2"
                     :: "v"(l0), "v"(goff), "s"(pbase) : "memory");
        asm volatile("global_load_async_to_lds_b32 %0, %1, %2"
                     :: "v"(l1), "v"(goff), "s"(tbase) : "memory");
    }
    asm volatile("s_wait_asynccnt 0x0" ::: "memory");
    __syncthreads();

    // ---- masks: clip(x*862-162, [-1024,700]) > -600 ; outside image -> 0 ----
    for (int idx = tid; idx < REG * REG; idx += 256) {
        int r  = idx / REG, c = idx % REG;
        int gy = y0 - 2 + r, gx = x0 - 2 + c;
        bool inb = (gy >= 0) & (gy < H) & (gx >= 0) & (gx < W);
        float ph = fminf(fmaxf(sf[0][idx] * 862.0f - 162.0f, -1024.0f), 700.0f);
        float th = fminf(fmaxf(sf[1][idx] * 862.0f - 162.0f, -1024.0f), 700.0f);
        msk[0][idx] = (unsigned char)((inb && (ph > -600.0f)) ? 1 : 0);
        msk[1][idx] = (unsigned char)((inb && (th > -600.0f)) ? 1 : 0);
    }
    if (tid < 3) red[tid] = 0.0f;
    __syncthreads();

    // ---- boundary = mask & ~erode3x3(mask), zero outside image ----
    for (int idx = tid; idx < BND * BND; idx += 256) {
        int r  = idx / BND, c = idx % BND;       // center at mask coords (r+1,c+1)
        int gy = y0 - 1 + r, gx = x0 - 1 + c;
        bool inb = (gy >= 0) & (gy < H) & (gx >= 0) & (gx < W);
        #pragma unroll
        for (int s = 0; s < 2; ++s) {
            unsigned m   = msk[s][(r + 1) * REG + (c + 1)];
            unsigned all = 1;
            #pragma unroll
            for (int dy = 0; dy < 3; ++dy)
                #pragma unroll
                for (int dx = 0; dx < 3; ++dx)
                    all &= msk[s][(r + dy) * REG + (c + dx)];
            bnd[s][idx] = (unsigned char)((inb && m && !all) ? 1 : 0);
        }
    }
    __syncthreads();

    // ---- surface = plus-dilation of boundary; accumulate counts ----
    int cI = 0, cP = 0, cT = 0;
    for (int idx = tid; idx < TILE * TILE; idx += 256) {   // exactly 4 iters, uniform
        int r = idx / TILE, c = idx % TILE;
        int br = r + 1, bc = c + 1;
        unsigned ps = bnd[0][br * BND + bc]
                    | bnd[0][(br - 1) * BND + bc] | bnd[0][(br + 1) * BND + bc]
                    | bnd[0][br * BND + bc - 1]   | bnd[0][br * BND + bc + 1];
        unsigned ts = bnd[1][br * BND + bc]
                    | bnd[1][(br - 1) * BND + bc] | bnd[1][(br + 1) * BND + bc]
                    | bnd[1][br * BND + bc - 1]   | bnd[1][br * BND + bc + 1];
        cP += (int)ps;
        cT += (int)ts;
        cI += (int)(ps & ts);
    }

    // ---- wave reduction via WMMA (uniform, EXEC all ones), then block/global ----
    float wI = wave_sum_wmma((float)cI);
    float wP = wave_sum_wmma((float)cP);
    float wT = wave_sum_wmma((float)cT);
    if ((tid & 31) == 0) {
        atomicAdd(&red[0], wI);
        atomicAdd(&red[1], wP);
        atomicAdd(&red[2], wT);
    }
    __syncthreads();
    if (tid == 0) {
        atomicAdd(&acc[b * 3 + 0], red[0]);
        atomicAdd(&acc[b * 3 + 1], red[1]);
        atomicAdd(&acc[b * 3 + 2], red[2]);
    }
}

__global__ void SurfaceDice_final(const float* __restrict__ acc,
                                  float* __restrict__ out) {
    int b = threadIdx.x;                 // 32 threads = 1 wave = 1 image each
    float inter = acc[b * 3 + 0];
    float sp    = acc[b * 3 + 1];
    float st    = acc[b * 3 + 2];
    float denom = sp + st;
    float dice  = (denom > 0.0f) ? (2.0f * inter / fmaxf(denom, 1.0f)) : 1.0f;
    #pragma unroll
    for (int o = 16; o > 0; o >>= 1) dice += __shfl_xor(dice, o, 32);
    if (b == 0) out[0] = 1.0f - dice / 32.0f;
}

extern "C" void kernel_launch(void* const* d_in, const int* in_sizes, int n_in,
                              void* d_out, int out_size, void* d_ws, size_t ws_size,
                              hipStream_t stream) {
    const float* pred = (const float*)d_in[0];
    const float* tgt  = (const float*)d_in[1];
    float* acc = (float*)d_ws;           // 96 floats: {inter, sumP, sumT} per image

    SurfaceDice_zero<<<1, NIMG * 3, 0, stream>>>(acc);
    dim3 grid(W / TILE, H / TILE, NIMG); // (16,16,32) = 8192 blocks
    SurfaceDice_main<<<grid, 256, 0, stream>>>(pred, tgt, acc);
    SurfaceDice_final<<<1, 32, 0, stream>>>(acc, (float*)d_out);
}
